// Attention_53953379172927
// MI455X (gfx1250) — matrix operand order repaired and checked by reading, not compiled, not run on hardware
//
#include <hip/hip_runtime.h>
#include <hip/hip_bf16.h>
#include <math.h>

// ---------------------------------------------------------------------------
// Problem constants (from reference)
// ---------------------------------------------------------------------------
#define B_      128
#define T_      2048
#define RNN_    1024
#define ATT_    128
#define ENC_    512
#define NF_     32
#define KS_     31
#define PAD_    15

typedef __attribute__((ext_vector_type(16))) _Float16 v16h;
typedef __attribute__((ext_vector_type(8)))  float    v8f;
typedef __attribute__((ext_vector_type(4)))  float    f32x4;

__device__ __forceinline__ float ntload(const float* p) {
    return __builtin_nontemporal_load(p);
}
__device__ __forceinline__ f32x4 ntload4(const f32x4* p) {
    return __builtin_nontemporal_load(p);
}
// Low 32 bits of a generic pointer to __shared__ are the LDS byte offset
// (shared aperture base occupies addr[63:32] per CDNA5 aperture rules).
__device__ __forceinline__ unsigned int lds_off(const void* p) {
    return (unsigned int)(size_t)p;
}

// ---------------------------------------------------------------------------
// Kernel 1: pq[b][a] = sum_r h[b][r] * qW[a][r]   (128x128x1024 GEMM, WMMA f16)
// One wave per 16x16 output tile; 8x8 tiles = 64 waves = 16 blocks x 4 waves.
// ---------------------------------------------------------------------------
__global__ __launch_bounds__(128)
void pq_gemm_kernel(const float* __restrict__ h,
                    const float* __restrict__ qW,
                    float* __restrict__ pq) {
    const int wave = (blockIdx.x * blockDim.x + threadIdx.x) >> 5;   // 0..63
    const int lane = threadIdx.x & 31;
    const int tb   = wave >> 3;          // b-tile 0..7
    const int ta   = wave & 7;           // a-tile 0..7
    const int half = lane >> 4;          // 0: lanes 0-15, 1: lanes 16-31
    const int l16  = lane & 15;

    const int m  = tb * 16 + l16;        // A row (b index) for this lane
    const int an = ta * 16 + l16;        // B col (a index) for this lane

    v8f acc = {};
    for (int k0 = 0; k0 < RNN_; k0 += 32) {
        // A fragment: 16-bit A 16x32 layout (ISA 7.12.2):
        //   lanes 0-15 : K = {k0+0..7, k0+16..23}
        //   lanes 16-31: K = {k0+8..15, k0+24..31}
        v16h A;
        const float* hrow = h + (size_t)m * RNN_ + k0 + half * 8;
        #pragma unroll
        for (int e = 0; e < 8; ++e) {
            A[e]     = (_Float16)hrow[e];
            A[e + 8] = (_Float16)hrow[e + 16];
        }
        // B fragment: B[k][n] = qW[n][k]; per lane N fixed, same K as A:
        //   lanes 0-15 : K = {k0+0..7, k0+16..23} ; lanes 16-31 : K = {k0+8..15, k0+24..31}
        v16h Bf;
        const float* wrow = qW + (size_t)an * RNN_ + k0 + half * 8;
        #pragma unroll
        for (int e = 0; e < 8; ++e) {
            Bf[e]     = (_Float16)wrow[e];
            Bf[e + 8] = (_Float16)wrow[e + 16];
        }

        acc = __builtin_amdgcn_wmma_f32_16x16x32_f16(
                false, A, false, Bf, (short)0, acc, false, false);
    }
    // D layout: VGPR j -> M = j (lanes 0-15) / 8+j (lanes 16-31), N = lane%16
    #pragma unroll
    for (int j = 0; j < 8; ++j) {
        int bb = tb * 16 + half * 8 + j;
        pq[bb * ATT_ + ta * 16 + l16] = acc[j];
    }
}

// ---------------------------------------------------------------------------
// Kernel 2: per-batch fused conv -> WMMA loc-projection -> tanh -> v-dot
//           -> masked softmax. One block (512 thr = 16 waves) per batch b.
// LDS ~66 KB -> 4 blocks/WGP = 64 waves/WGP (full occupancy) to hide the
// latency of the 134 MB processed_memory stream.
// ---------------------------------------------------------------------------
#define CHUNK_T 256
#define LOCP    33          // padded f16 row stride for loc_sh

__global__ __launch_bounds__(512)
void attn_energy_softmax_kernel(const float* __restrict__ pq,
                                const float* __restrict__ pm,
                                const float* __restrict__ awc,
                                const int*   __restrict__ mask,
                                const float* __restrict__ cW,
                                const float* __restrict__ lW,
                                const float* __restrict__ vW,
                                float* __restrict__ w_out) {
    const int b    = blockIdx.x;
    const int tid  = threadIdx.x;          // 0..511
    const int wave = tid >> 5;             // 0..15
    const int lane = tid & 31;
    const int half = lane >> 4;
    const int l16  = lane & 15;

    __shared__ float    aw0[T_ + 2 * PAD_ + 2];     // zero-padded channel 0
    __shared__ float    aw1[T_ + 2 * PAD_ + 2];     // zero-padded channel 1
    __shared__ float    cw_sh[NF_ * 2 * KS_];       // conv weights
    __shared__ _Float16 lw_sh[ATT_ * NF_];          // loc_W in f16
    __shared__ _Float16 loc_sh[CHUNK_T * LOCP];     // conv output chunk (f16)
    __shared__ float    e_part[8 * CHUNK_T];        // per-a-tile partial sums
    __shared__ float    e_sh[T_];                   // energies
    __shared__ float    pq_sh[ATT_];
    __shared__ float    v_sh[ATT_];
    __shared__ float    red[17];

    // ---- stage inputs into LDS -------------------------------------------
    for (int i = tid; i < T_ + 2 * PAD_; i += 512) {
        int ti = i - PAD_;
        float a0 = 0.f, a1 = 0.f;
        if (ti >= 0 && ti < T_) {
            a0 = awc[((size_t)b * 2 + 0) * T_ + ti];
            a1 = awc[((size_t)b * 2 + 1) * T_ + ti];
        }
        aw0[i] = a0; aw1[i] = a1;
    }
    for (int i = tid; i < NF_ * 2 * KS_; i += 512) cw_sh[i] = cW[i];
    for (int i = tid; i < ATT_ * NF_;   i += 512) lw_sh[i] = (_Float16)lW[i];
    if (tid < ATT_) { pq_sh[tid] = pq[b * ATT_ + tid]; v_sh[tid] = vW[tid]; }
    __syncthreads();

    // Per-wave constants: this wave owns a-tile (wave&7)
    const int at = wave & 7;
    const int a  = at * 16 + l16;          // this lane's a index
    // B fragment: B[k=f][n=a] = loc_W[a][f]; lanes 0-15: f={0..7,16..23}, 16-31: f={8..15,24..31}
    v16h Bf;
    #pragma unroll
    for (int e = 0; e < 8; ++e) {
        Bf[e]     = lw_sh[a * NF_ + half * 8 + e];
        Bf[e + 8] = lw_sh[a * NF_ + half * 8 + 16 + e];
    }
    const float vv  = v_sh[a];
    const float pqv = pq_sh[a];
    const int fbase = half * 8;            // A fragment f offsets per half-wave

    const float* pm_b = pm + (size_t)b * T_ * ATT_;

    // ---- main loop over 256-t chunks -------------------------------------
    for (int c0 = 0; c0 < T_; c0 += CHUNK_T) {
        // conv phase: loc[t][f], computed once per chunk by whole block
        for (int idx = tid; idx < CHUNK_T * NF_; idx += 512) {
            int tl = idx >> 5;             // local t
            int f  = idx & (NF_ - 1);
            int t  = c0 + tl;
            const float* wf = &cw_sh[f * 2 * KS_];
            float accv = 0.f;
            #pragma unroll
            for (int k = 0; k < KS_; ++k)
                accv += aw0[t + k] * wf[k] + aw1[t + k] * wf[KS_ + k];
            loc_sh[tl * LOCP + f] = (_Float16)accv;
        }
        __syncthreads();

        // WMMA phase: 16 t-strips x 8 a-tiles = 128 tiles, 8 per wave
        const int ttl0 = wave >> 3;        // 0 or 1
        for (int s = 0; s < 8; ++s) {
            const int ttl = ttl0 + s * 2;  // strip 0..15 within chunk
            const int tl  = ttl * 16 + l16;

            // pm loads issued FIRST (independent of WMMA) so the 8 global
            // loads overlap the matrix op instead of serializing after it.
            float pmv[8];
            #pragma unroll
            for (int j = 0; j < 8; ++j) {
                int tloc = ttl * 16 + (half << 3) + j;     // 0..255
                pmv[j] = ntload(pm_b + (size_t)(c0 + tloc) * ATT_ + a);
            }

            // A fragment: A[m=t][k=f] = loc[t][f]
            v16h A;
            #pragma unroll
            for (int e = 0; e < 8; ++e) {
                A[e]     = loc_sh[tl * LOCP + fbase + e];
                A[e + 8] = loc_sh[tl * LOCP + fbase + 16 + e];
            }
            v8f acc = {};
            acc = __builtin_amdgcn_wmma_f32_16x16x32_f16(
                    false, A, false, Bf, (short)0, acc, false, false);

            // epilogue: + pq + pm, tanh, *v, reduce over a (16 lanes/half)
            #pragma unroll
            for (int j = 0; j < 8; ++j) {
                int tloc = ttl * 16 + (half << 3) + j;
                float sv = tanhf(acc[j] + pqv + pmv[j]) * vv;
                #pragma unroll
                for (int off = 8; off; off >>= 1)
                    sv += __shfl_xor(sv, off, 32);
                if (l16 == 0) e_part[at * CHUNK_T + tloc] = sv;
            }
        }
        __syncthreads();

        // deterministic reduce over the 8 a-tiles
        for (int tl = tid; tl < CHUNK_T; tl += 512) {
            float s = 0.f;
            #pragma unroll
            for (int q = 0; q < 8; ++q) s += e_part[q * CHUNK_T + tl];
            e_sh[c0 + tl] = s;
        }
        __syncthreads();
    }

    // ---- masked softmax over T --------------------------------------------
    const float NEG_INF = -__builtin_inff();
    float m = NEG_INF;
    for (int t = tid; t < T_; t += 512) {
        float e = e_sh[t];
        if (mask[(size_t)b * T_ + t] != 0) e = NEG_INF;
        e_sh[t] = e;
        m = fmaxf(m, e);
    }
    #pragma unroll
    for (int off = 16; off; off >>= 1) m = fmaxf(m, __shfl_xor(m, off, 32));
    if (lane == 0) red[wave] = m;
    __syncthreads();
    if (tid == 0) {
        float mm = red[0];
        #pragma unroll
        for (int q = 1; q < 16; ++q) mm = fmaxf(mm, red[q]);
        red[16] = mm;
    }
    __syncthreads();
    const float M = red[16];
    __syncthreads();

    float z = 0.f;
    for (int t = tid; t < T_; t += 512) {
        float p = __expf(e_sh[t] - M);     // exp(-inf)=0 for masked
        e_sh[t] = p;
        z += p;
    }
    #pragma unroll
    for (int off = 16; off; off >>= 1) z += __shfl_xor(z, off, 32);
    if (lane == 0) red[wave] = z;
    __syncthreads();
    if (tid == 0) {
        float zz = 0.f;
        #pragma unroll
        for (int q = 0; q < 16; ++q) zz += red[q];
        red[16] = (zz > 0.f) ? zz : 1.f;
    }
    __syncthreads();
    const float inv = 1.f / red[16];
    for (int t = tid; t < T_; t += 512)
        w_out[(size_t)b * T_ + t] = e_sh[t] * inv;
}

// ---------------------------------------------------------------------------
// Kernel 3: context partials. grid (B, 8 t-splits) x 128 thr; each thread owns
// one float4 column; streams memory rows fully coalesced (2 KB / row / block).
// This kernel carries 80% of all HBM traffic (537 MB) -> NT loads + L2
// prefetch 16 rows (32 KB) ahead; weights staged via async-to-LDS (ASYNCcnt).
// ---------------------------------------------------------------------------
__global__ __launch_bounds__(128)
void ctx_partial_kernel(const float* __restrict__ mem,
                        const float* __restrict__ w,
                        float* __restrict__ part) {
    const int b   = blockIdx.x;
    const int ts  = blockIdx.y;        // 0..7
    const int tid = threadIdx.x;       // 0..127
    __shared__ float w_sh[256];
    const int t0 = ts * 256;

    // Stage 256 softmax weights into LDS with gfx1250 async-to-LDS copies
    // (tracked by ASYNCcnt, no VGPR round trip).
    {
        const float* wbase = w + (size_t)b * T_ + t0;   // uniform -> SGPR pair
        unsigned int voff0 = (unsigned int)(tid * 4);
        unsigned int voff1 = (unsigned int)((tid + 128) * 4);
        unsigned int l0 = lds_off(&w_sh[tid]);
        unsigned int l1 = lds_off(&w_sh[tid + 128]);
        asm volatile("global_load_async_to_lds_b32 %0, %1, %2"
                     :: "v"(l0), "v"(voff0), "s"(wbase) : "memory");
        asm volatile("global_load_async_to_lds_b32 %0, %1, %2"
                     :: "v"(l1), "v"(voff1), "s"(wbase) : "memory");
        asm volatile("s_wait_asynccnt 0x0" ::: "memory");
    }
    __syncthreads();

    const f32x4* m4 = (const f32x4*)(mem + (size_t)b * T_ * ENC_);
    f32x4 acc = {0.f, 0.f, 0.f, 0.f};
    #pragma unroll 8
    for (int t = 0; t < 256; ++t) {
        // prefetch 16 rows ahead; one lane per 128B cacheline
        if ((tid & 7) == 0 && t + 16 < 256)
            __builtin_prefetch(
                (const void*)&m4[(size_t)(t0 + t + 16) * (ENC_ / 4) + tid], 0, 0);
        f32x4 v = ntload4(&m4[(size_t)(t0 + t) * (ENC_ / 4) + tid]);
        acc += v * w_sh[t];
    }
    f32x4* p4 = (f32x4*)(part + ((size_t)b * 8 + ts) * ENC_);
    p4[tid] = acc;
}

// ---------------------------------------------------------------------------
// Kernel 4: deterministic reduction of the 8 partials -> context output.
// ---------------------------------------------------------------------------
__global__ __launch_bounds__(512)
void ctx_reduce_kernel(const float* __restrict__ part,
                       float* __restrict__ outc) {
    const int b = blockIdx.x;
    const int d = threadIdx.x;         // 0..511
    float s = 0.f;
    #pragma unroll
    for (int j = 0; j < 8; ++j)
        s += part[((size_t)b * 8 + j) * ENC_ + d];
    outc[(size_t)b * ENC_ + d] = s;
}

// ---------------------------------------------------------------------------
extern "C" void kernel_launch(void* const* d_in, const int* in_sizes, int n_in,
                              void* d_out, int out_size, void* d_ws, size_t ws_size,
                              hipStream_t stream) {
    const float* h    = (const float*)d_in[0];   // [128,1024]
    const float* mem  = (const float*)d_in[1];   // [128,2048,512]
    const float* pm   = (const float*)d_in[2];   // [128,2048,128]
    const float* awc  = (const float*)d_in[3];   // [128,2,2048]
    const int*   mask = (const int*)  d_in[4];   // [128,2048] (bool as int)
    const float* qW   = (const float*)d_in[5];   // [128,1024]
    const float* cW   = (const float*)d_in[6];   // [32,2,31]
    const float* lW   = (const float*)d_in[7];   // [128,32]
    const float* vW   = (const float*)d_in[8];   // [1,128]

    float* outC = (float*)d_out;                  // [128,512]
    float* outW = (float*)d_out + B_ * ENC_;      // [128,2048]

    float* ws   = (float*)d_ws;
    float* pq   = ws;                             // 128*128 floats (64 KB)
    float* part = ws + B_ * ATT_;                 // 128*8*512 floats (2 MB)

    pq_gemm_kernel<<<16, 128, 0, stream>>>(h, qW, pq);
    attn_energy_softmax_kernel<<<B_, 512, 0, stream>>>(pq, pm, awc, mask,
                                                       cW, lW, vW, outW);
    ctx_partial_kernel<<<dim3(B_, 8), 128, 0, stream>>>(mem, outW, part);
    ctx_reduce_kernel<<<B_, 512, 0, stream>>>(part, outC);
}
